// Imp_8993661518660
// MI455X (gfx1250) — matrix operand order repaired and checked by reading, not compiled, not run on hardware
//
#include <hip/hip_runtime.h>

// ---------------------------------------------------------------------------
// Problem constants (reference: B=1, N_SUP=16384, N_QRY=16384, C=1024, D=1024)
// ---------------------------------------------------------------------------
#define N_SUP 16384
#define N_QRY 16384
#define C_DIM 1024
#define D_DIM 1024
#define LOG_2PI 1.8378770664093453f

#define TILE   128   // block tile (M and N)
#define KT     32    // K step per WMMA (bf16 16x16x32)
#define LDS_RS 36    // fp32 LDS row stride for [row][k] tiles (144B, 16B-aligned)

typedef __attribute__((ext_vector_type(16))) __bf16 v16bf;
typedef __attribute__((ext_vector_type(2)))  __bf16 v2bf;
typedef __attribute__((ext_vector_type(8)))  float  v8f;
typedef __attribute__((ext_vector_type(4)))  unsigned int u32x4;
typedef __attribute__((ext_vector_type(8)))  int  i32x8;
typedef __attribute__((ext_vector_type(4)))  int  i32x4;

// --------------------------- feature detection -----------------------------
#ifndef USE_TDM
#ifdef __has_builtin
#if __has_builtin(__builtin_amdgcn_tensor_load_to_lds) && \
    __has_builtin(__builtin_amdgcn_s_wait_tensorcnt)
#define USE_TDM 1
#endif
#endif
#endif
#ifndef USE_TDM
#define USE_TDM 0
#endif

#ifndef HAVE_PK_CVT
#ifdef __has_builtin
#if __has_builtin(__builtin_amdgcn_cvt_pk_bf16_f32)
#define HAVE_PK_CVT 1
#endif
#endif
#endif
#ifndef HAVE_PK_CVT
#define HAVE_PK_CVT 0
#endif

// Convert a pair of floats to packed bf16 (RNE) using the native converter
// when available (single v_cvt_pk_bf16_f32, co-executes with WMMA).
__device__ __forceinline__ v2bf pk_bf16(float a, float b) {
#if HAVE_PK_CVT
  return __builtin_amdgcn_cvt_pk_bf16_f32(a, b);
#else
  v2bf r;
  r[0] = static_cast<__bf16>(a);
  r[1] = static_cast<__bf16>(b);
  return r;
#endif
}

__device__ __forceinline__ v8f wmma_bf16(v16bf a, v16bf b, v8f c) {
  // 8 args: (neg_a, A, neg_b, B, c_mod, C, reuse_a, reuse_b)
  return __builtin_amdgcn_wmma_f32_16x16x32_bf16(false, a, false, b,
                                                 (short)0, c, false, false);
}

#if USE_TDM
// LDS byte offset of a __shared__ pointer (addrspace(3) ptrtoint)
#define AS3 __attribute__((address_space(3)))
__device__ __forceinline__ unsigned lds_off(void* p) {
  return (unsigned)(unsigned long long)(AS3 void*)p;
}

// Issue a 2D TENSOR_LOAD_TO_LDS: copy a tile_y x tile_x (rows x contiguous
// elems) fp32 tile whose rows are row_stride elements apart in global memory
// into LDS starting at lds_addr, optionally inserting pad DWORDs so the LDS
// row stride matches the fragment-load layout.
// D# bitfields per CDNA5 ISA 8.3/8.4 (group0 128b, group1 256b).
__device__ __forceinline__ void tdm_load_2d(const void* g, unsigned lds_addr,
                                            unsigned tile_x, unsigned tile_y,
                                            unsigned long long row_stride,
                                            unsigned pad_en, unsigned pad_int,
                                            unsigned pad_amt) {
  unsigned long long ga = (unsigned long long)g;
  u32x4 g0;
  g0[0] = 1u;                                    // count=1, user mode
  g0[1] = lds_addr;                              // lds_addr [63:32]
  g0[2] = (unsigned)ga;                          // global_addr lo
  g0[3] = (unsigned)((ga >> 32) & 0x1FFFFFFull)  // global_addr [56:32]
          | 0x80000000u;                         // type=2 ("image") [127:126]
  i32x8 g1;
  g1[0] = (int)((2u << 16)                       // data_size=2 (4B)
                | (pad_en << 20) | (pad_int << 22) | (pad_amt << 25));
  g1[1] = (int)((tile_x & 0xFFFFu) << 16);       // tensor_dim0 lo16 -> [63:48]
  g1[2] = (int)((tile_x >> 16) | ((tile_y & 0xFFFFu) << 16)); // dim0 hi, dim1 lo
  g1[3] = (int)((tile_y >> 16) | (tile_x << 16));             // dim1 hi, tile_dim0
  g1[4] = (int)tile_y;                           // tile_dim1 (tile_dim2=0)
  g1[5] = (int)(unsigned)row_stride;             // tensor_dim0_stride lo32
  g1[6] = (int)(unsigned)(row_stride >> 32);     // stride hi16 (dim1_stride=0)
  g1[7] = 0;
  i32x4 z4 = {0, 0, 0, 0};
#if __clang_major__ >= 23
  i32x8 z8 = {0, 0, 0, 0, 0, 0, 0, 0};
  __builtin_amdgcn_tensor_load_to_lds(g0, g1, z4, z4, z8, 0);
#else
  __builtin_amdgcn_tensor_load_to_lds(g0, g1, z4, z4, 0);
#endif
}
#endif  // USE_TDM

// ------------------- fragment loaders (fp32 LDS -> bf16) -------------------
// A fragment from row-major LDS tile [row][k], stride LDS_RS:
// lane l: M=(l&15), K = kb..kb+7 -> elems 0..7, kb+16..kb+23 -> elems 8..15,
// kb = (l>>4)*8.
__device__ __forceinline__ v16bf frag_a_rm(const float* s, int row, int lane) {
  const int kb = (lane >> 4) << 3;
  const float* p = s + row * LDS_RS + kb;
  union { v16bf v; v2bf p2[8]; } u;
  #pragma unroll
  for (int i = 0; i < 4; ++i) {
    u.p2[i]     = pk_bf16(p[2 * i], p[2 * i + 1]);
    u.p2[4 + i] = pk_bf16(p[16 + 2 * i], p[17 + 2 * i]);
  }
  return u.v;
}
// B fragment from row-major LDS tile [col][k]: lane l: N=(l&15),
// K = kb..kb+7 -> elems 0..7, kb+16..kb+23 -> elems 8..15, kb = (l>>4)*8.
__device__ __forceinline__ v16bf frag_b_rm(const float* s, int col, int lane) {
  const int kb = (lane >> 4) << 3;
  const float* p = s + col * LDS_RS + kb;
  union { v16bf v; v2bf p2[8]; } u;
  #pragma unroll
  for (int i = 0; i < 8; ++i) u.p2[i] = pk_bf16(p[2 * i + (i >= 4 ? 8 : 0)], p[2 * i + 1 + (i >= 4 ? 8 : 0)]);
  return u.v;
}
// Same fragments from a k-major LDS tile [k][m] (stride TILE), used when the
// global layout is transposed and the TDM copied natural rows.
__device__ __forceinline__ v16bf frag_a_km(const float* s, int m, int lane) {
  const int kb = (lane >> 4) << 3;
  union { v16bf v; v2bf p2[8]; } u;
  #pragma unroll
  for (int i = 0; i < 4; ++i) {
    u.p2[i]     = pk_bf16(s[(kb + 2 * i) * TILE + m], s[(kb + 2 * i + 1) * TILE + m]);
    u.p2[4 + i] = pk_bf16(s[(kb + 16 + 2 * i) * TILE + m],
                          s[(kb + 17 + 2 * i) * TILE + m]);
  }
  return u.v;
}
__device__ __forceinline__ v16bf frag_b_km(const float* s, int n, int lane) {
  const int kb = (lane >> 4) << 3;
  union { v16bf v; v2bf p2[8]; } u;
  #pragma unroll
  for (int i = 0; i < 8; ++i)
    u.p2[i] = pk_bf16(s[(kb + 2 * i + (i >= 4 ? 8 : 0)) * TILE + n], s[(kb + 2 * i + 1 + (i >= 4 ? 8 : 0)) * TILE + n]);
  return u.v;
}

// ---------------------------------------------------------------------------
// Kernel 1: inv_prob_sum[c] = 1 / (sum_n probs[n,c]  with zero->1 guard)
// ---------------------------------------------------------------------------
__global__ __launch_bounds__(256) void k_colsum(const float* __restrict__ probs,
                                                float* __restrict__ inv_sum) {
  int c = blockIdx.x * blockDim.x + threadIdx.x;
  if (c >= C_DIM) return;
  float s = 0.f;
  for (int n = 0; n < N_SUP; ++n) s += probs[(size_t)n * C_DIM + c];
  if (s == 0.f) s = 1.f;
  inv_sum[c] = 1.f / s;
}

// ---------------------------------------------------------------------------
// Kernel 2: protos[c,d] = inv_sum[c] * sum_n probs[n,c] * h_train[n,d]
// M=c, N=d, K=n.  Global layout is [n][c]/[n][d] (K-major), so the TDM copies
// natural [k][m] rows (512B bursts) and fragment loads gather strided.
// ---------------------------------------------------------------------------
__global__ __launch_bounds__(256) void k_gemm_protos(
    const float* __restrict__ probs,    // [N_SUP, C]
    const float* __restrict__ h_train,  // [N_SUP, D]
    const float* __restrict__ inv_sum,  // [C]
    float* __restrict__ protos)         // [C, D]
{
#if USE_TDM
  __shared__ __align__(16) float As[2][KT * TILE];  // [k][m]
  __shared__ __align__(16) float Bs[2][KT * TILE];
#else
  __shared__ __align__(16) float As[1][KT * TILE];
  __shared__ __align__(16) float Bs[1][KT * TILE];
#endif
  const int m0 = blockIdx.y * TILE;     // c block
  const int n0 = blockIdx.x * TILE;     // d block
  const int t = threadIdx.x;
  const int lane = t & 31;
  const int w = t >> 5;
  const int wm = w & 1;                 // 2 waves along M (64 rows each)
  const int wn = w >> 1;                // 4 waves along N (32 cols each)
  const int NT = N_SUP / KT;

  const v8f vzero = {0.f, 0.f, 0.f, 0.f, 0.f, 0.f, 0.f, 0.f};
  v8f acc[4][2];
  #pragma unroll
  for (int i = 0; i < 4; ++i)
    #pragma unroll
    for (int j = 0; j < 2; ++j) acc[i][j] = vzero;

#if USE_TDM
  if (w == 0) {
    tdm_load_2d(probs + m0,   lds_off(&As[0][0]), TILE, KT, C_DIM, 0, 0, 0);
    tdm_load_2d(h_train + n0, lds_off(&Bs[0][0]), TILE, KT, D_DIM, 0, 0, 0);
  }
#endif

  for (int it = 0; it < NT; ++it) {
#if USE_TDM
    const int cur = it & 1;
    if (w == 0) __builtin_amdgcn_s_wait_tensorcnt(0);  // tile `it` landed
    __syncthreads();  // publish LDS; prior compute on the other buffer done
    if (w == 0 && it + 1 < NT) {                       // overlap next DMA
      const int nxt = (it + 1) & 1;
      const size_t kg = (size_t)(it + 1) * KT;
      tdm_load_2d(probs + kg * C_DIM + m0,   lds_off(&As[nxt][0]),
                  TILE, KT, C_DIM, 0, 0, 0);
      tdm_load_2d(h_train + kg * D_DIM + n0, lds_off(&Bs[nxt][0]),
                  TILE, KT, D_DIM, 0, 0, 0);
    }
    const float* as = As[cur];
    const float* bs = Bs[cur];
#else
    const int k0 = it * KT;
    #pragma unroll
    for (int j = 0; j < 16; ++j) {
      int idx = t + j * 256;
      int k = idx >> 7;
      int m = idx & 127;
      As[0][k * TILE + m] = probs[(size_t)(k0 + k) * C_DIM + (m0 + m)];
      Bs[0][k * TILE + m] = h_train[(size_t)(k0 + k) * D_DIM + (n0 + m)];
    }
    __syncthreads();
    const float* as = As[0];
    const float* bs = Bs[0];
#endif

    v16bf af[4], bfg[2];
    #pragma unroll
    for (int i = 0; i < 4; ++i)
      af[i] = frag_a_km(as, wm * 64 + i * 16 + (lane & 15), lane);
    #pragma unroll
    for (int j = 0; j < 2; ++j)
      bfg[j] = frag_b_km(bs, wn * 32 + j * 16 + (lane & 15), lane);

    #pragma unroll
    for (int i = 0; i < 4; ++i)
      #pragma unroll
      for (int j = 0; j < 2; ++j)
        acc[i][j] = wmma_bf16(af[i], bfg[j], acc[i][j]);

#if !USE_TDM
    __syncthreads();
#endif
  }

  // D-matrix layout: m = r + 8*(lane>>4), n = lane&15 per 16x16 tile
  #pragma unroll
  for (int i = 0; i < 4; ++i) {
    #pragma unroll
    for (int j = 0; j < 2; ++j) {
      const int col  = n0 + wn * 32 + j * 16 + (lane & 15);
      const int rowb = m0 + wm * 64 + i * 16 + ((lane >> 4) << 3);
      #pragma unroll
      for (int r = 0; r < 8; ++r) {
        const int row = rowb + r;
        protos[(size_t)row * D_DIM + col] = acc[i][j][r] * inv_sum[row];
      }
    }
  }
}

// ---------------------------------------------------------------------------
// Kernel 3: row sum-of-squares (c2 over protos, x2 over h_test)
// ---------------------------------------------------------------------------
__global__ __launch_bounds__(256) void k_rownorm(const float* __restrict__ x,
                                                 float* __restrict__ out) {
  const int row = blockIdx.x;
  const float* p = x + (size_t)row * D_DIM;
  float s = 0.f;
  for (int i = threadIdx.x; i < D_DIM; i += 256) {
    float v = p[i];
    s += v * v;
  }
  __shared__ float red[256];
  red[threadIdx.x] = s;
  __syncthreads();
  #pragma unroll
  for (int off = 128; off > 0; off >>= 1) {
    if (threadIdx.x < off) red[threadIdx.x] += red[threadIdx.x + off];
    __syncthreads();
  }
  if (threadIdx.x == 0) out[row] = red[0];
}

// ---------------------------------------------------------------------------
// Kernel 4: logits[m,c] = (2*xc - x2[m] - c2[c]) / (2*radii[c])
//                         - 0.5*D*(log(radii[c]) + LOG_2PI),  xc = h_test.protos
// Both operands are K-contiguous row-major; the TDM pad feature produces the
// 36-float padded LDS row stride directly (32 DWORDs data + 4 DWORDs pad).
// ---------------------------------------------------------------------------
__global__ __launch_bounds__(256) void k_gemm_logits(
    const float* __restrict__ h_test,  // [M, D]
    const float* __restrict__ protos,  // [C, D]
    const float* __restrict__ x2,      // [M]
    const float* __restrict__ c2,      // [C]
    const float* __restrict__ radii,   // [C]
    float* __restrict__ out)           // [M, C]
{
#if USE_TDM
  __shared__ __align__(16) float As[2][TILE * LDS_RS];  // [m][k], padded rows
  __shared__ __align__(16) float Bs[2][TILE * LDS_RS];
#else
  __shared__ __align__(16) float As[1][TILE * LDS_RS];
  __shared__ __align__(16) float Bs[1][TILE * LDS_RS];
#endif
  const int m0 = blockIdx.y * TILE;   // query block
  const int n0 = blockIdx.x * TILE;   // cluster block
  const int t = threadIdx.x;
  const int lane = t & 31;
  const int w = t >> 5;
  const int wm = w & 1;
  const int wn = w >> 1;
  const int NT = D_DIM / KT;

  const v8f vzero = {0.f, 0.f, 0.f, 0.f, 0.f, 0.f, 0.f, 0.f};
  v8f acc[4][2];
  #pragma unroll
  for (int i = 0; i < 4; ++i)
    #pragma unroll
    for (int j = 0; j < 2; ++j) acc[i][j] = vzero;

#if USE_TDM
  // pad_int=4 -> pad every 32 DWORDs; pad_amt=3 -> 4 DWORDs pad (stride 36)
  if (w == 0) {
    tdm_load_2d(h_test + (size_t)m0 * D_DIM, lds_off(&As[0][0]),
                KT, TILE, D_DIM, 1, 4, 3);
    tdm_load_2d(protos + (size_t)n0 * D_DIM, lds_off(&Bs[0][0]),
                KT, TILE, D_DIM, 1, 4, 3);
  }
#endif

  for (int it = 0; it < NT; ++it) {
#if USE_TDM
    const int cur = it & 1;
    if (w == 0) __builtin_amdgcn_s_wait_tensorcnt(0);
    __syncthreads();
    if (w == 0 && it + 1 < NT) {
      const int nxt = (it + 1) & 1;
      const int kg = (it + 1) * KT;
      tdm_load_2d(h_test + (size_t)m0 * D_DIM + kg, lds_off(&As[nxt][0]),
                  KT, TILE, D_DIM, 1, 4, 3);
      tdm_load_2d(protos + (size_t)n0 * D_DIM + kg, lds_off(&Bs[nxt][0]),
                  KT, TILE, D_DIM, 1, 4, 3);
    }
    const float* as = As[cur];
    const float* bs = Bs[cur];
#else
    const int k0 = it * KT;
    #pragma unroll
    for (int j = 0; j < 16; ++j) {
      int idx = t + j * 256;
      int k = idx & 31;
      int m = idx >> 5;
      As[0][m * LDS_RS + k] = h_test[(size_t)(m0 + m) * D_DIM + k0 + k];
      Bs[0][m * LDS_RS + k] = protos[(size_t)(n0 + m) * D_DIM + k0 + k];
    }
    __syncthreads();
    const float* as = As[0];
    const float* bs = Bs[0];
#endif

    v16bf af[4], bfg[2];
    #pragma unroll
    for (int i = 0; i < 4; ++i)
      af[i] = frag_a_rm(as, wm * 64 + i * 16 + (lane & 15), lane);
    #pragma unroll
    for (int j = 0; j < 2; ++j)
      bfg[j] = frag_b_rm(bs, wn * 32 + j * 16 + (lane & 15), lane);

    #pragma unroll
    for (int i = 0; i < 4; ++i)
      #pragma unroll
      for (int j = 0; j < 2; ++j)
        acc[i][j] = wmma_bf16(af[i], bfg[j], acc[i][j]);

#if !USE_TDM
    __syncthreads();
#endif
  }

  #pragma unroll
  for (int i = 0; i < 4; ++i) {
    #pragma unroll
    for (int j = 0; j < 2; ++j) {
      const int col = n0 + wn * 32 + j * 16 + (lane & 15);
      const float rad = radii[col];
      const float invr = 0.5f / rad;                      // 1/(2*radii)
      const float nc = 0.5f * (float)D_DIM * (__logf(rad) + LOG_2PI);
      const float c2v = c2[col];
      const int rowb = m0 + wm * 64 + i * 16 + ((lane >> 4) << 3);
      #pragma unroll
      for (int r = 0; r < 8; ++r) {
        const int row = rowb + r;
        const float xc = acc[i][j][r];
        out[(size_t)row * C_DIM + col] = (2.f * xc - x2[row] - c2v) * invr - nc;
      }
    }
  }
}

// ---------------------------------------------------------------------------
// Host-side launch
// ---------------------------------------------------------------------------
extern "C" void kernel_launch(void* const* d_in, const int* in_sizes, int n_in,
                              void* d_out, int out_size, void* d_ws, size_t ws_size,
                              hipStream_t stream) {
  (void)in_sizes; (void)n_in; (void)out_size; (void)ws_size;
  const float* h_train    = (const float*)d_in[0];  // [N_SUP, D]
  const float* prob_train = (const float*)d_in[1];  // [N_SUP, C]
  const float* radii      = (const float*)d_in[2];  // [C]
  const float* h_test     = (const float*)d_in[3];  // [N_QRY, D]
  float* out = (float*)d_out;                       // [N_QRY, C]

  float* protos  = (float*)d_ws;                    // C*D
  float* inv_sum = protos + (size_t)C_DIM * D_DIM;  // C
  float* c2      = inv_sum + C_DIM;                 // C
  float* x2      = c2 + C_DIM;                      // N_QRY

  k_colsum<<<(C_DIM + 255) / 256, 256, 0, stream>>>(prob_train, inv_sum);

  dim3 g1(D_DIM / TILE, C_DIM / TILE);              // (8, 8)
  k_gemm_protos<<<g1, 256, 0, stream>>>(prob_train, h_train, inv_sum, protos);

  k_rownorm<<<C_DIM, 256, 0, stream>>>(protos, c2);
  k_rownorm<<<N_QRY, 256, 0, stream>>>(h_test, x2);

  dim3 g2(C_DIM / TILE, N_QRY / TILE);              // (8, 128)
  k_gemm_logits<<<g2, 256, 0, stream>>>(h_test, protos, x2, c2, radii, out);
}
